// SelfAttention_61881888801231
// MI455X (gfx1250) — matrix operand order repaired and checked by reading, not compiled, not run on hardware
//
#include <hip/hip_runtime.h>
#include <math.h>

// Problem constants
#define B_   2
#define T_   2048
#define C_   1024
#define H_   16
#define D_   64
#define EPS_ 1e-6f

typedef _Float16 f16;
typedef __attribute__((ext_vector_type(16))) _Float16 v16h;
typedef __attribute__((ext_vector_type(8)))  _Float16 v8h;
typedef __attribute__((ext_vector_type(8)))  float    v8f;

union V16 { v16h v; v8h h[2]; };

// ---- async global->LDS (ASYNCcnt path), guarded: falls back to sync copy ----
#if defined(__has_builtin)
#  if __has_builtin(__builtin_amdgcn_global_load_async_to_lds_b128) && \
      __has_builtin(__builtin_amdgcn_s_wait_asynccnt)
#    define HAVE_ASYNC 1
#  endif
#endif
#ifndef HAVE_ASYNC
#  define HAVE_ASYNC 0
#endif

// Builtin signature (from hipcc diagnostic): param1 is
//   'int __vector_size__(4*sizeof(int)) __device__ *'  -> 4xi32 in AS1
typedef int v4i_g __attribute__((vector_size(16)));
typedef __attribute__((address_space(1))) v4i_g gv4i;   // global 4xi32
typedef __attribute__((address_space(3))) v4i_g lv4i;   // LDS    4xi32

__device__ __forceinline__ void copy_b128(const f16* g, f16* l) {
#if HAVE_ASYNC
    __builtin_amdgcn_global_load_async_to_lds_b128(
        (gv4i*)(uintptr_t)g,
        (lv4i*)(uint32_t)(uintptr_t)l, 0, 0);
#else
    *(float4*)l = *(const float4*)g;
#endif
}

__device__ __forceinline__ void wait_async_copies() {
#if HAVE_ASYNC
    __builtin_amdgcn_s_wait_asynccnt(0);
#endif
}

// Split-precision helper: x = (float)hi + (float)lo to ~2^-21 relative.
__device__ __forceinline__ void split2(float x, f16& h, f16& l) {
    h = (f16)x;
    l = (f16)(x - (float)h);
}

// A-fragment (16x32 f16): lane half picks K = {8*half..+8} and {16+8*half..+8}
__device__ __forceinline__ v16h load_a_frag(const f16* row, int chunk, int half) {
    V16 u;
    u.h[0] = *(const v8h*)(row + chunk + 8 * half);
    u.h[1] = *(const v8h*)(row + chunk + 16 + 8 * half);
    return u.v;
}
// B-fragment (32x16 f16): lane half picks K = {8*half..+8} and {16+8*half..+8}
__device__ __forceinline__ v16h load_b_frag(const f16* row, int chunk, int half) {
    V16 u;
    u.h[0] = *(const v8h*)(row + chunk + 8 * half);
    u.h[1] = *(const v8h*)(row + chunk + 16 + 8 * half);
    return u.v;
}

// ---------------------------------------------------------------------------
// row sum-of-squares (one wave per row)
// ---------------------------------------------------------------------------
__global__ __launch_bounds__(256) void rowsq_kernel(const float* __restrict__ X,
                                                    float* __restrict__ out, int K) {
    const int wave = threadIdx.x >> 5;
    const int lane = threadIdx.x & 31;
    const int row  = blockIdx.x * 8 + wave;
    const float* p = X + (size_t)row * K;
    float s = 0.0f;
    for (int i = lane; i < K; i += 32) { float v = p[i]; s += v * v; }
#pragma unroll
    for (int m = 16; m >= 1; m >>= 1) s += __shfl_xor(s, m, 32);
    if (lane == 0) out[row] = s;
}

// ---------------------------------------------------------------------------
// column sum-of-squares
// ---------------------------------------------------------------------------
__global__ __launch_bounds__(256) void colsq_kernel(const float* __restrict__ W,
                                                    float* __restrict__ out, int K, int N) {
    const int n = blockIdx.x * blockDim.x + threadIdx.x;
    if (n >= N) return;
    float s = 0.0f;
    for (int r = 0; r < K; ++r) { float v = W[(size_t)r * N + n]; s += v * v; }
    out[n] = s;
}

// ---------------------------------------------------------------------------
// elementwise f32 -> (hi,lo) f16 split, same layout
// ---------------------------------------------------------------------------
__global__ __launch_bounds__(256) void split_kernel(const float* __restrict__ X,
                                                    f16* __restrict__ Xh,
                                                    f16* __restrict__ Xl, int n) {
    const int idx = blockIdx.x * blockDim.x + threadIdx.x;
    if (idx >= n) return;
    f16 h, l; split2(X[idx], h, l);
    Xh[idx] = h; Xl[idx] = l;
}

// ---------------------------------------------------------------------------
// f32 (K,N) -> (hi,lo) f16 (N,K) transpose split (one-time weight prep)
// ---------------------------------------------------------------------------
__global__ __launch_bounds__(256) void split_transpose_kernel(const float* __restrict__ W,
                                                              f16* __restrict__ Wth,
                                                              f16* __restrict__ Wtl,
                                                              int K, int N) {
    const size_t idx = (size_t)blockIdx.x * blockDim.x + threadIdx.x;
    if (idx >= (size_t)K * N) return;
    const int k = (int)(idx / N);
    const int n = (int)(idx % N);
    f16 h, l; split2(W[idx], h, l);
    Wth[(size_t)n * K + k] = h;
    Wtl[(size_t)n * K + k] = l;
}

// ---------------------------------------------------------------------------
// yat-dense GEMM on V_WMMA_F32_16X16X32_F16, split-precision (3 products).
//   A: (M,K) hi/lo f16 planes; B: (N,K) transposed hi/lo f16 planes.
//   block = 128 (4 waves), tile 64x64, K-chunks of 32; async tile staging.
// ---------------------------------------------------------------------------
#define GK  32
#define AST 40   // LDS stride (f16), padded: 16 distinct bank groups for b128 frags

__global__ __launch_bounds__(128) void yat_gemm_f16_kernel(
    const f16* __restrict__ Ahi, const f16* __restrict__ Alo,    // M x K
    const f16* __restrict__ Bth, const f16* __restrict__ Btl,    // N x K
    const float* __restrict__ bias, const float* __restrict__ alpha,
    const float* __restrict__ xsq, const float* __restrict__ ksq,
    float* __restrict__ OUT, int M, int N, int K)
{
    __shared__ f16 lAh[64 * AST], lAl[64 * AST];
    __shared__ f16 lBh[64 * AST], lBl[64 * AST];

    const int tid  = threadIdx.x;
    const int wave = tid >> 5;
    const int lane = tid & 31;
    const int half = lane >> 4;
    const int l16  = lane & 15;
    const int m0   = blockIdx.y * 64;
    const int n0   = blockIdx.x * 64;

    v8f acc[4] = {v8f{}, v8f{}, v8f{}, v8f{}};

    const int r   = tid >> 1;          // 0..63
    const int seg = (tid & 1) * 16;    // 0 or 16

    for (int k0 = 0; k0 < K; k0 += GK) {
        // stage A (64x32) and B (64x32) hi/lo: 2x b128 per thread per buffer
        {
            const size_t ga = (size_t)(m0 + r) * K + k0 + seg;
            copy_b128(&Ahi[ga],     &lAh[r * AST + seg]);
            copy_b128(&Ahi[ga + 8], &lAh[r * AST + seg + 8]);
            copy_b128(&Alo[ga],     &lAl[r * AST + seg]);
            copy_b128(&Alo[ga + 8], &lAl[r * AST + seg + 8]);
            const size_t gb = (size_t)(n0 + r) * K + k0 + seg;
            copy_b128(&Bth[gb],     &lBh[r * AST + seg]);
            copy_b128(&Bth[gb + 8], &lBh[r * AST + seg + 8]);
            copy_b128(&Btl[gb],     &lBl[r * AST + seg]);
            copy_b128(&Btl[gb + 8], &lBl[r * AST + seg + 8]);
        }
        wait_async_copies();
        __syncthreads();

        const v16h ah = load_a_frag(&lAh[(wave * 16 + l16) * AST], 0, half);
        const v16h al = load_a_frag(&lAl[(wave * 16 + l16) * AST], 0, half);
#pragma unroll
        for (int j = 0; j < 4; ++j) {
            const v16h bh = load_b_frag(&lBh[(j * 16 + l16) * AST], 0, half);
            const v16h bl = load_b_frag(&lBl[(j * 16 + l16) * AST], 0, half);
            acc[j] = __builtin_amdgcn_wmma_f32_16x16x32_f16(false, ah, false, bh, (short)0, acc[j], false, false);
            acc[j] = __builtin_amdgcn_wmma_f32_16x16x32_f16(false, ah, false, bl, (short)0, acc[j], false, false);
            acc[j] = __builtin_amdgcn_wmma_f32_16x16x32_f16(false, al, false, bh, (short)0, acc[j], false, false);
        }
        __syncthreads();
    }

    // yat epilogue: y = dot^2 / (|x|^2 + |k|^2 - 2 dot + eps) * scale + bias
    const float of = (float)N;
    const float sc = powf(sqrtf(of) / logf(1.0f + of), alpha[0]);
#pragma unroll
    for (int j = 0; j < 4; ++j) {
        const int n    = n0 + j * 16 + l16;
        const float kq = ksq[n];
        const float bv = bias[n];
#pragma unroll
        for (int i = 0; i < 8; ++i) {
            const int m      = m0 + wave * 16 + half * 8 + i;
            const float dot  = acc[j][i];
            const float dist = xsq[m] + kq - 2.0f * dot + EPS_;
            OUT[(size_t)m * N + n] = (dot * dot / dist) * sc + bv;
        }
    }
}

// ---------------------------------------------------------------------------
// rotary + transpose + split:
//   qkv (B,T,3,H,D) f32 -> q,k hi/lo (B,H,T,D) f16 ; v hi/lo transposed (B,H,D,T)
// ---------------------------------------------------------------------------
__global__ __launch_bounds__(256) void rotary_split_kernel(
    const float* __restrict__ qkv,
    f16* __restrict__ qh, f16* __restrict__ ql,
    f16* __restrict__ kh, f16* __restrict__ kl,
    f16* __restrict__ vth, f16* __restrict__ vtl)
{
    const int idx = blockIdx.x * blockDim.x + threadIdx.x;
    const int j   = idx & 31;          // D/2 = 32
    int rest      = idx >> 5;
    const int h   = rest & (H_ - 1);
    rest >>= 4;
    const int t   = rest & (T_ - 1);
    const int b   = rest >> 11;
    const int bh  = b * H_ + h;

    const size_t base = (size_t)(b * T_ + t) * (3 * C_);
    const size_t dst  = ((size_t)bh * T_ + t) * D_;

    const float inv_freq = powf(10000.0f, -((float)(2 * j) / (float)D_));
    const float ang = (float)t * inv_freq;
    const float sn = sinf(ang), cs = cosf(ang);

    f16 hh, ll;
    // q (slot 0)
    {
        const float x1 = qkv[base + (size_t)h * D_ + j];
        const float x2 = qkv[base + (size_t)h * D_ + j + 32];
        split2(x1 * cs - x2 * sn, hh, ll); qh[dst + j] = hh;      ql[dst + j] = ll;
        split2(x2 * cs + x1 * sn, hh, ll); qh[dst + j + 32] = hh; ql[dst + j + 32] = ll;
    }
    // k (slot 1)
    {
        const float x1 = qkv[base + C_ + (size_t)h * D_ + j];
        const float x2 = qkv[base + C_ + (size_t)h * D_ + j + 32];
        split2(x1 * cs - x2 * sn, hh, ll); kh[dst + j] = hh;      kl[dst + j] = ll;
        split2(x2 * cs + x1 * sn, hh, ll); kh[dst + j + 32] = hh; kl[dst + j + 32] = ll;
    }
    // v (slot 2): transpose to (B,H,D,T)
    {
        const size_t vb = (size_t)bh * D_ * T_;
        const float v1 = qkv[base + 2 * C_ + (size_t)h * D_ + j];
        const float v2 = qkv[base + 2 * C_ + (size_t)h * D_ + j + 32];
        split2(v1, hh, ll); vth[vb + (size_t)j * T_ + t] = hh;        vtl[vb + (size_t)j * T_ + t] = ll;
        split2(v2, hh, ll); vth[vb + (size_t)(j + 32) * T_ + t] = hh; vtl[vb + (size_t)(j + 32) * T_ + t] = ll;
    }
}

// ---------------------------------------------------------------------------
// flash attention (causal) on V_WMMA_F32_16X16X32_F16; async K/V staging.
//   S = QK^T: split x split (3 products); O += P V: P(hi) x split V (2 products)
//   grid = (T/64, B*H), block = 128 (4 waves); wave w owns query rows w*16..+16
// ---------------------------------------------------------------------------
#define QST 72   // padded LDS stride (f16)

__global__ __launch_bounds__(128) void attn_f16_kernel(
    const f16* __restrict__ Qhi, const f16* __restrict__ Qlo,   // (B,H,T,D)
    const f16* __restrict__ Khi, const f16* __restrict__ Klo,   // (B,H,T,D)
    const f16* __restrict__ Vth, const f16* __restrict__ Vtl,   // (B,H,D,T)
    float* __restrict__ Oout)                                   // (B,T,C)
{
    __shared__ f16 lQh[64 * QST], lQl[64 * QST];
    __shared__ f16 lKh[64 * QST], lKl[64 * QST];  // lKh reused as P staging
    __shared__ f16 lVh[64 * QST], lVl[64 * QST];

    const int tid  = threadIdx.x;
    const int wave = tid >> 5;
    const int lane = tid & 31;
    const int half = lane >> 4;
    const int l16  = lane & 15;
    const int qt   = blockIdx.x;
    const int bh   = blockIdx.y;
    const int b    = bh >> 4;
    const int h    = bh & 15;

    const f16* Qhb = Qhi + (size_t)bh * T_ * D_;
    const f16* Qlb = Qlo + (size_t)bh * T_ * D_;
    const f16* Khb = Khi + (size_t)bh * T_ * D_;
    const f16* Klb = Klo + (size_t)bh * T_ * D_;
    const f16* Vhb = Vth + (size_t)bh * D_ * T_;
    const f16* Vlb = Vtl + (size_t)bh * D_ * T_;

    const int r   = tid >> 1;        // 0..63
    const int seg = (tid & 1) * 32;  // 0 or 32

    // stage Q tile (rows qt*64..+64): 4x b128 per thread per buffer
    {
        const size_t g = (size_t)(qt * 64 + r) * D_ + seg;
#pragma unroll
        for (int i = 0; i < 4; ++i) {
            copy_b128(&Qhb[g + i * 8], &lQh[r * QST + seg + i * 8]);
            copy_b128(&Qlb[g + i * 8], &lQl[r * QST + seg + i * 8]);
        }
    }

    v8f o[4] = {v8f{}, v8f{}, v8f{}, v8f{}};
    float mrow[8], lrow[8];
#pragma unroll
    for (int i = 0; i < 8; ++i) { mrow[i] = -1e30f; lrow[i] = 0.0f; }

    const float inv_scale = 0.125f;  // 1/sqrt(64)

    for (int kt = 0; kt <= qt; ++kt) {
        __syncthreads();  // previous P / V fully consumed by all waves
        // stage K rows and V^T rows for this key tile (async)
        {
            const size_t gk = (size_t)(kt * 64 + r) * D_ + seg;   // K: (T,D) rows
            const size_t gv = (size_t)r * T_ + kt * 64 + seg;     // V^T: (D,T) rows
#pragma unroll
            for (int i = 0; i < 4; ++i) {
                copy_b128(&Khb[gk + i * 8], &lKh[r * QST + seg + i * 8]);
                copy_b128(&Klb[gk + i * 8], &lKl[r * QST + seg + i * 8]);
                copy_b128(&Vhb[gv + i * 8], &lVh[r * QST + seg + i * 8]);
                copy_b128(&Vlb[gv + i * 8], &lVl[r * QST + seg + i * 8]);
            }
        }
        wait_async_copies();   // covers Q staging on first iteration too
        __syncthreads();

        // ---- S = Q K^T (16x64 per wave), K-dim = d, 2 chunks of 32 ----
        v8f s[4] = {v8f{}, v8f{}, v8f{}, v8f{}};
#pragma unroll
        for (int c = 0; c < 2; ++c) {
            const v16h ah = load_a_frag(&lQh[(wave * 16 + l16) * QST], c * 32, half);
            const v16h al = load_a_frag(&lQl[(wave * 16 + l16) * QST], c * 32, half);
#pragma unroll
            for (int j = 0; j < 4; ++j) {
                const v16h bhf = load_b_frag(&lKh[(j * 16 + l16) * QST], c * 32, half);
                const v16h blf = load_b_frag(&lKl[(j * 16 + l16) * QST], c * 32, half);
                s[j] = __builtin_amdgcn_wmma_f32_16x16x32_f16(false, ah, false, bhf, (short)0, s[j], false, false);
                s[j] = __builtin_amdgcn_wmma_f32_16x16x32_f16(false, ah, false, blf, (short)0, s[j], false, false);
                s[j] = __builtin_amdgcn_wmma_f32_16x16x32_f16(false, al, false, bhf, (short)0, s[j], false, false);
            }
        }

        // ---- scale + causal mask ----
        const int rowbase = qt * 64 + wave * 16 + half * 8;
#pragma unroll
        for (int j = 0; j < 4; ++j) {
            const int col = kt * 64 + j * 16 + l16;
#pragma unroll
            for (int i = 0; i < 8; ++i) {
                float sv = s[j][i] * inv_scale;
                if (col > rowbase + i) sv = -1e30f;
                s[j][i] = sv;
            }
        }

        // ---- online softmax (row reductions inside 16-lane halves) ----
        float mnew[8], corr[8];
#pragma unroll
        for (int i = 0; i < 8; ++i) {
            float vmax = fmaxf(fmaxf(s[0][i], s[1][i]), fmaxf(s[2][i], s[3][i]));
#pragma unroll
            for (int msk = 1; msk < 16; msk <<= 1)
                vmax = fmaxf(vmax, __shfl_xor(vmax, msk, 32));
            mnew[i] = fmaxf(vmax, mrow[i]);
            corr[i] = __expf(mrow[i] - mnew[i]);
        }
#pragma unroll
        for (int i = 0; i < 8; ++i) {
            float rs = 0.0f;
#pragma unroll
            for (int j = 0; j < 4; ++j) {
                float p = __expf(s[j][i] - mnew[i]);
                s[j][i] = p;
                rs += p;
            }
#pragma unroll
            for (int msk = 1; msk < 16; msk <<= 1)
                rs += __shfl_xor(rs, msk, 32);
            lrow[i] = lrow[i] * corr[i] + rs;
            mrow[i] = mnew[i];
        }
#pragma unroll
        for (int j = 0; j < 4; ++j)
#pragma unroll
            for (int i = 0; i < 8; ++i) o[j][i] *= corr[i];

        __syncthreads();  // all waves done with lKh/lKl B-fragment reads

        // stage P (hi f16) into lKh: own wave's 16 rows, row-major [q][key]
#pragma unroll
        for (int j = 0; j < 4; ++j)
#pragma unroll
            for (int i = 0; i < 8; ++i)
                lKh[(wave * 16 + half * 8 + i) * QST + j * 16 + l16] = (f16)s[j][i];

        // ---- O += P V (K-dim = keys, 2 chunks of 32; same-wave LDS in-order) ----
#pragma unroll
        for (int c = 0; c < 2; ++c) {
            const v16h p = load_a_frag(&lKh[(wave * 16 + l16) * QST], c * 32, half);
#pragma unroll
            for (int j = 0; j < 4; ++j) {
                const v16h vh = load_b_frag(&lVh[(j * 16 + l16) * QST], c * 32, half);
                const v16h vl = load_b_frag(&lVl[(j * 16 + l16) * QST], c * 32, half);
                o[j] = __builtin_amdgcn_wmma_f32_16x16x32_f16(false, p, false, vh, (short)0, o[j], false, false);
                o[j] = __builtin_amdgcn_wmma_f32_16x16x32_f16(false, p, false, vl, (short)0, o[j], false, false);
            }
        }
    }

    // ---- normalize + store (B,T,C) ----
    float invl[8];
#pragma unroll
    for (int i = 0; i < 8; ++i) invl[i] = 1.0f / lrow[i];
#pragma unroll
    for (int j = 0; j < 4; ++j) {
#pragma unroll
        for (int i = 0; i < 8; ++i) {
            const int trow = qt * 64 + wave * 16 + half * 8 + i;
            Oout[((size_t)(b * T_ + trow)) * C_ + h * D_ + j * 16 + l16] = o[j][i] * invl[i];
        }
    }
}

// ---------------------------------------------------------------------------
// host-side orchestration
// ---------------------------------------------------------------------------
extern "C" void kernel_launch(void* const* d_in, const int* in_sizes, int n_in,
                              void* d_out, int out_size, void* d_ws, size_t ws_size,
                              hipStream_t stream) {
    const float* x      = (const float*)d_in[0];
    // d_in[1] = mask (tril; recomputed analytically in-kernel)
    const float* w_attn = (const float*)d_in[2];
    const float* b_attn = (const float*)d_in[3];
    const float* a_attn = (const float*)d_in[4];
    const float* w_proj = (const float*)d_in[5];
    const float* b_proj = (const float*)d_in[6];
    const float* a_proj = (const float*)d_in[7];
    float* out = (float*)d_out;

    const int M  = B_ * T_;       // 4096
    const int N1 = 3 * C_;        // 3072
    const int N2 = C_;            // 1024
    const int K  = C_;            // 1024
    const size_t SZ = (size_t)B_ * H_ * T_ * D_;  // 4,194,304

    char* cur = (char*)d_ws;
    auto alloc = [&](size_t bytes) -> void* {
        void* p = (void*)cur;
        cur += (bytes + 255) & ~(size_t)255;
        return p;
    };

    float* qkv   = (float*)alloc((size_t)M * N1 * 4);
    f16*   xhi   = (f16*)alloc((size_t)M * K * 2);
    f16*   xlo   = (f16*)alloc((size_t)M * K * 2);
    f16*   wta_h = (f16*)alloc((size_t)N1 * K * 2);
    f16*   wta_l = (f16*)alloc((size_t)N1 * K * 2);
    f16*   wtp_h = (f16*)alloc((size_t)N2 * K * 2);
    f16*   wtp_l = (f16*)alloc((size_t)N2 * K * 2);
    f16*   qh    = (f16*)alloc(SZ * 2);
    f16*   ql    = (f16*)alloc(SZ * 2);
    f16*   kh    = (f16*)alloc(SZ * 2);
    f16*   kl    = (f16*)alloc(SZ * 2);
    f16*   vth   = (f16*)alloc(SZ * 2);
    f16*   vtl   = (f16*)alloc(SZ * 2);
    float* attn  = (float*)alloc((size_t)M * C_ * 4);
    f16*   oh    = (f16*)alloc((size_t)M * K * 2);
    f16*   ol    = (f16*)alloc((size_t)M * K * 2);
    float* xsq_x = (float*)alloc((size_t)M * 4);
    float* xsq_o = (float*)alloc((size_t)M * 4);
    float* ksq_a = (float*)alloc((size_t)N1 * 4);
    float* ksq_p = (float*)alloc((size_t)N2 * 4);

    // norms
    rowsq_kernel<<<M / 8, 256, 0, stream>>>(x, xsq_x, K);
    colsq_kernel<<<(N1 + 255) / 256, 256, 0, stream>>>(w_attn, ksq_a, K, N1);
    colsq_kernel<<<(N2 + 255) / 256, 256, 0, stream>>>(w_proj, ksq_p, K, N2);

    // one-time split/transpose prep
    split_kernel<<<(M * K) / 256, 256, 0, stream>>>(x, xhi, xlo, M * K);
    split_transpose_kernel<<<(K * N1) / 256, 256, 0, stream>>>(w_attn, wta_h, wta_l, K, N1);
    split_transpose_kernel<<<(K * N2) / 256, 256, 0, stream>>>(w_proj, wtp_h, wtp_l, K, N2);

    // qkv = yat_dense(x, w_attn)
    yat_gemm_f16_kernel<<<dim3(N1 / 64, M / 64), 128, 0, stream>>>(
        xhi, xlo, wta_h, wta_l, b_attn, a_attn, xsq_x, ksq_a, qkv, M, N1, K);

    // rotary + transpose + split
    rotary_split_kernel<<<(B_ * T_ * H_ * (D_ / 2)) / 256, 256, 0, stream>>>(
        qkv, qh, ql, kh, kl, vth, vtl);

    // flash attention
    attn_f16_kernel<<<dim3(T_ / 64, B_ * H_), 128, 0, stream>>>(
        qh, ql, kh, kl, vth, vtl, attn);

    // out = yat_dense(attn, w_proj)
    rowsq_kernel<<<M / 8, 256, 0, stream>>>(attn, xsq_o, K);
    split_kernel<<<(M * C_) / 256, 256, 0, stream>>>(attn, oh, ol, M * C_);
    yat_gemm_f16_kernel<<<dim3(N2 / 64, M / 64), 128, 0, stream>>>(
        oh, ol, wtp_h, wtp_l, b_proj, a_proj, xsq_o, ksq_p, out, M, N2, K);

    (void)in_sizes; (void)n_in; (void)out_size; (void)ws_size;
}
